// GroupedQueryAttention_30296699306037
// MI455X (gfx1250) — matrix operand order repaired and checked by reading, not compiled, not run on hardware
//
#include <hip/hip_runtime.h>
#include <hip/hip_bf16.h>

typedef __attribute__((ext_vector_type(16))) _Float16 v16h;
typedef __attribute__((ext_vector_type(8)))  _Float16 v8h;
typedef __attribute__((ext_vector_type(4)))  _Float16 v4h;
typedef __attribute__((ext_vector_type(8)))  float    v8f;
typedef __attribute__((ext_vector_type(4)))  float    v4f;

#define S_LEN   4096
#define H_DIM   1024
#define KV_DIM  256
#define HEAD_D  64
#define N_HEADS 16

// 1/sqrt(64) * log2(e): scores come out ready for exp2
#define Q_SCALE 0.18033688011112042f

// Raw hardware transcendentals: softmax weights below 2^-126 are zero anyway,
// so the libm denormal-range fixup around v_exp_f32 is pure overhead here.
#if __has_builtin(__builtin_amdgcn_exp2f)
#define EXP2F(x) __builtin_amdgcn_exp2f(x)
#else
#define EXP2F(x) exp2f(x)
#endif
#if __has_builtin(__builtin_amdgcn_rcpf)
#define RCPF(x) __builtin_amdgcn_rcpf(x)
#else
#define RCPF(x) (1.0f / (x))
#endif

// ---------------------------------------------------------------------------
// A-fragment loader (16-bit A 16x32), contiguous K to match B:
//   lanes 0-15  (row = lane)    hold K = kb+0..15
//   lanes 16-31 (row = lane-16) hold K = kb+16..31
__device__ inline v16h load_a_frag(const _Float16* row, int kb) {
    const int kh = (threadIdx.x >> 4) & 1;
    return *(const v16h*)(row + kb + kh * 16);
}

// B-fragment (16-bit B 32x16): lane = output column n (lane&15),
// lanes 0-15 hold K = kb+0..15, lanes 16-31 hold K = kb+16..31 (contiguous).
__device__ inline v16h load_b_frag(const _Float16* col, int kb) {
    const int kh = (threadIdx.x >> 4) & 1;
    return *(const v16h*)(col + kb + kh * 16);
}

// ---------------------------------------------------------------------------
// 16-lane max reduction with DPP (VALU-only; no LDS, no DScnt waits).
template <int CTRL>
__device__ inline float fmax_dpp(float x) {
    int s = __builtin_amdgcn_update_dpp(0, __float_as_int(x), CTRL, 0xF, 0xF, true);
    return fmaxf(x, __int_as_float(s));
}
__device__ inline float rowmax16(float x) {
    x = fmax_dpp<0xB1>(x);    // quad_perm [1,0,3,2]  : xor 1
    x = fmax_dpp<0x4E>(x);    // quad_perm [2,3,0,1]  : xor 2
    x = fmax_dpp<0x141>(x);   // row_half_mirror      : combine quads in 8-group
    x = fmax_dpp<0x140>(x);   // row_mirror           : combine 8-groups in row
    return x;
}

// ---------------------------------------------------------------------------
// f32 -> f16 conversion, 4 elements / thread
__global__ __launch_bounds__(256)
void cvt_f32_f16(const float* __restrict__ src, _Float16* __restrict__ dst, int n4) {
    int i = blockIdx.x * 256 + threadIdx.x;
    if (i < n4) {
        v4f f = ((const v4f*)src)[i];
        v4h h;
        h[0] = (_Float16)f[0]; h[1] = (_Float16)f[1];
        h[2] = (_Float16)f[2]; h[3] = (_Float16)f[3];
        ((v4h*)dst)[i] = h;
    }
}

// ---------------------------------------------------------------------------
// Projection GEMM: Out[m,n] = (sum_k X[m,k]*W[n,k] + bias[n]) * out_scale
// One wave computes a 32x64 output tile (two A-frags share each B-frag).
// TR=1 stores Out transposed [N,M] (V^T for the attention kernel).
template <bool TR>
__global__ __launch_bounds__(32)
void proj_wmma(const _Float16* __restrict__ X, const _Float16* __restrict__ W,
               const float* __restrict__ bias, _Float16* __restrict__ Out,
               int M, int N, int K, float out_scale) {
    const int lane = threadIdx.x;
    const int ln   = lane & 15;
    const int h16  = lane >> 4;
    const int m0   = blockIdx.y * 32;
    const int n0   = blockIdx.x * 64;

    const _Float16* xrow0 = X + (size_t)(m0 + ln) * K;
    const _Float16* xrow1 = X + (size_t)(m0 + 16 + ln) * K;
    v8f c[2][4] = {};

    for (int kk = 0; kk < K; kk += 32) {
        __builtin_prefetch(xrow0 + kk + 256, 0, 1);   // global_prefetch_b8
        __builtin_prefetch(xrow1 + kk + 256, 0, 1);
        v16h a0 = load_a_frag(xrow0, kk);
        v16h a1 = load_a_frag(xrow1, kk);
#pragma unroll
        for (int j = 0; j < 4; ++j) {
            const _Float16* wrow = W + (size_t)(n0 + j * 16 + ln) * K;
            v16h b = load_b_frag(wrow, kk);
            c[0][j] = __builtin_amdgcn_wmma_f32_16x16x32_f16(
                false, a0, false, b, (short)0, c[0][j], false, false);
            c[1][j] = __builtin_amdgcn_wmma_f32_16x16x32_f16(
                false, a1, false, b, (short)0, c[1][j], false, false);
        }
    }

#pragma unroll
    for (int i = 0; i < 2; ++i) {
#pragma unroll
        for (int j = 0; j < 4; ++j) {
            const int n  = n0 + j * 16 + ln;
            const float bv = bias[n];
            if (TR) {
                // rows v+8*h16 consecutive in v -> one contiguous v8h store
                v8h st;
#pragma unroll
                for (int v = 0; v < 8; ++v)
                    st[v] = (_Float16)((c[i][j][v] + bv) * out_scale);
                *(v8h*)(Out + (size_t)n * M + m0 + i * 16 + h16 * 8) = st;
            } else {
#pragma unroll
                for (int v = 0; v < 8; ++v)
                    Out[(size_t)(m0 + i * 16 + v + 8 * h16) * N + n] =
                        (_Float16)((c[i][j][v] + bv) * out_scale);
            }
        }
    }
}

// ---------------------------------------------------------------------------
// score tile: two 16x16 C-frags of Q.K^T for keys [t0, t0+32)
// k0/k1 are per-lane row pointers hoisted out of the key loop.
__device__ inline void score_tile(const _Float16* k0, const _Float16* k1, int t0,
                                  const v16h aq[2], v8f sc[2]) {
    const size_t off = (size_t)t0 * KV_DIM;
#pragma unroll
    for (int dk = 0; dk < 2; ++dk) {
        v16h b0 = load_b_frag(k0 + off, dk * 32);
        sc[0] = __builtin_amdgcn_wmma_f32_16x16x32_f16(
            false, aq[dk], false, b0, (short)0, sc[0], false, false);
        v16h b1 = load_b_frag(k1 + off, dk * 32);
        sc[1] = __builtin_amdgcn_wmma_f32_16x16x32_f16(
            false, aq[dk], false, b1, (short)0, sc[1], false, false);
    }
}

// ---------------------------------------------------------------------------
// Flash attention: one wave per (head, 16-query-row tile).
// Qh: [S,1024] f16 pre-scaled by 1/sqrt(d)*log2(e)  -> exp2-ready scores
// Kh: [S,256]  f16 ; Vt: [256,S] f16 (transposed V) ; Out: [S,1024] f32
// Denominator = P x ones via WMMA (same C-frag layout as O: zero shuffles).
// Next tile's scores are computed between softmax and P.V to hide the
// WMMA->VALU hazard slots and the LDS P-relayout latency.
__global__ __launch_bounds__(32)
void attn_wmma(const _Float16* __restrict__ Qh, const _Float16* __restrict__ Kh,
               const _Float16* __restrict__ Vt, float* __restrict__ Out) {
    __shared__ _Float16 ldsP[16 * 32];

    const int lane = threadIdx.x;
    const int ln   = lane & 15;
    const int h16  = lane >> 4;
    const int head = blockIdx.y;           // 0..15
    const int kv   = head >> 2;            // shared KV head
    const int kvoff = kv * HEAD_D;
    const int s0   = blockIdx.x * 16;

    // Q tile as two A-fragments (d = 0..31, 32..63)
    const _Float16* qrow = Qh + (size_t)(s0 + ln) * H_DIM + head * HEAD_D;
    v16h aq[2];
    aq[0] = load_a_frag(qrow, 0);
    aq[1] = load_a_frag(qrow, 32);

    v16h ones;
#pragma unroll
    for (int i = 0; i < 16; ++i) ones[i] = (_Float16)1.0f;

    // hoisted per-lane base pointers for K rows and V^T rows
    const _Float16* k0 = Kh + (size_t)ln * KV_DIM + kvoff;
    const _Float16* k1 = k0 + (size_t)16 * KV_DIM;
    const _Float16* vbase = Vt + (size_t)(kvoff + ln) * S_LEN + h16 * 16;

    float mi[8];
#pragma unroll
    for (int v = 0; v < 8; ++v) mi[v] = -3.0e30f;
    v8f o[4] = {};
    v8f osum = {};                          // running softmax denominator

    v8f sc[2] = {};
    score_tile(k0, k1, 0, aq, sc);          // pipeline prologue

    for (int t0 = 0; t0 < S_LEN; t0 += 32) {
        // ---- online softmax on current tile (DPP row-max, raw exp2) ----
#pragma unroll
        for (int v = 0; v < 8; ++v) {
            float mx   = rowmax16(fmaxf(sc[0][v], sc[1][v]));
            float mn   = fmaxf(mi[v], mx);
            float corr = EXP2F(mi[v] - mn);     // bare v_exp_f32
            mi[v] = mn;

            float p0 = EXP2F(sc[0][v] - mn);
            float p1 = EXP2F(sc[1][v] - mn);
            // P re-layout via LDS: C-frag (row v+8*h16, col ln / ln+16)
            ldsP[(v + 8 * h16) * 32 + ln]      = (_Float16)p0;
            ldsP[(v + 8 * h16) * 32 + 16 + ln] = (_Float16)p1;

            osum[v] *= corr;
            o[0][v] *= corr; o[1][v] *= corr; o[2][v] *= corr; o[3][v] *= corr;
        }

        // ---- overlap: next tile's score WMMAs (independent of LDS/P.V) ----
        v8f scn[2] = {};
        if (t0 + 32 < S_LEN)
            score_tile(k0, k1, t0 + 32, aq, scn);

        // ---- O += P V ; denominator += P x ones (same A-frag) ----
        const _Float16* prow = &ldsP[ln * 32];
        v16h ap = load_a_frag(prow, 0);      // same-wave ds ordering via DScnt
        osum = __builtin_amdgcn_wmma_f32_16x16x32_f16(
            false, ap, false, ones, (short)0, osum, false, false);
#pragma unroll
        for (int j = 0; j < 4; ++j) {
            v16h bv = *(const v16h*)(vbase + (size_t)j * 16 * S_LEN + t0);
            o[j] = __builtin_amdgcn_wmma_f32_16x16x32_f16(
                false, ap, false, bv, (short)0, o[j], false, false);
        }

        sc[0] = scn[0];
        sc[1] = scn[1];
    }

    // ---- normalize (one raw rcp per row) and store f32 output ----
#pragma unroll
    for (int v = 0; v < 8; ++v) {
        const float inv = RCPF(osum[v]);
#pragma unroll
        for (int j = 0; j < 4; ++j) {
            Out[(size_t)(s0 + v + 8 * h16) * H_DIM + head * HEAD_D + j * 16 + ln] =
                o[j][v] * inv;
        }
    }
}

// ---------------------------------------------------------------------------
extern "C" void kernel_launch(void* const* d_in, const int* in_sizes, int n_in,
                              void* d_out, int out_size, void* d_ws, size_t ws_size,
                              hipStream_t stream) {
    (void)in_sizes; (void)n_in; (void)out_size; (void)ws_size;

    const float* x   = (const float*)d_in[0];   // [4096,1024]
    const float* q_w = (const float*)d_in[1];   // [1024,1024]
    const float* q_b = (const float*)d_in[2];   // [1024]
    const float* k_w = (const float*)d_in[3];   // [256,1024]
    const float* k_b = (const float*)d_in[4];   // [256]
    const float* v_w = (const float*)d_in[5];   // [256,1024]
    const float* v_b = (const float*)d_in[6];   // [256]
    float* out = (float*)d_out;                 // [4096,1024]

    // workspace layout (halves)
    _Float16* ws  = (_Float16*)d_ws;
    _Float16* Xh  = ws;                                   // 4096*1024
    _Float16* QW  = Xh + (size_t)S_LEN * H_DIM;           // 1024*1024
    _Float16* KW  = QW + (size_t)H_DIM * H_DIM;           // 256*1024
    _Float16* VW  = KW + (size_t)KV_DIM * H_DIM;          // 256*1024
    _Float16* Qh  = VW + (size_t)KV_DIM * H_DIM;          // 4096*1024
    _Float16* Kh  = Qh + (size_t)S_LEN * H_DIM;           // 4096*256
    _Float16* Vt  = Kh + (size_t)S_LEN * KV_DIM;          // 256*4096 (transposed)

    // 1) f32 -> f16 conversions
    cvt_f32_f16<<<(S_LEN * H_DIM / 4 + 255) / 256, 256, 0, stream>>>(x,   Xh, S_LEN * H_DIM / 4);
    cvt_f32_f16<<<(H_DIM * H_DIM / 4 + 255) / 256, 256, 0, stream>>>(q_w, QW, H_DIM * H_DIM / 4);
    cvt_f32_f16<<<(KV_DIM * H_DIM / 4 + 255) / 256, 256, 0, stream>>>(k_w, KW, KV_DIM * H_DIM / 4);
    cvt_f32_f16<<<(KV_DIM * H_DIM / 4 + 255) / 256, 256, 0, stream>>>(v_w, VW, KV_DIM * H_DIM / 4);

    // 2) WMMA projections (one wave per 32x64 tile)
    dim3 blk(32, 1, 1);
    proj_wmma<false><<<dim3(H_DIM  / 64, S_LEN / 32), blk, 0, stream>>>(Xh, QW, q_b, Qh, S_LEN, H_DIM,  H_DIM, Q_SCALE);
    proj_wmma<false><<<dim3(KV_DIM / 64, S_LEN / 32), blk, 0, stream>>>(Xh, KW, k_b, Kh, S_LEN, KV_DIM, H_DIM, 1.0f);
    proj_wmma<true ><<<dim3(KV_DIM / 64, S_LEN / 32), blk, 0, stream>>>(Xh, VW, v_b, Vt, S_LEN, KV_DIM, H_DIM, 1.0f);

    // 3) flash attention, one wave per (head, 16 query rows)
    attn_wmma<<<dim3(S_LEN / 16, N_HEADS), blk, 0, stream>>>(Qh, Kh, Vt, out);
}
